// Attention_84928683311352
// MI455X (gfx1250) — matrix operand order repaired and checked by reading, not compiled, not run on hardware
//
#include <hip/hip_runtime.h>
#include <hip/hip_bf16.h>

typedef __attribute__((ext_vector_type(16))) _Float16 v16h;
typedef __attribute__((ext_vector_type(8)))  _Float16 h8;
typedef __attribute__((ext_vector_type(4)))  _Float16 h4;
typedef __attribute__((ext_vector_type(8)))  float    v8f;

#define NUM_SEQS       2
#define LQ             1024
#define NUM_HEADS      32
#define NUM_KV_HEADS   8
#define GQA_GROUP      (NUM_HEADS / NUM_KV_HEADS)
#define HEAD_DIM       128
#define BLOCK_SIZE     64
#define MAX_NUM_BLOCKS 32
#define QTILE          128           // q rows per workgroup (8 waves x 16)
#define KTILE          64            // keys per iteration (== BLOCK_SIZE)
#define NWAVES         8
#define THREADS        256
// softmax scale folded with log2(e): scores land directly in exp2 domain
#define QSCALE_LOG2    (0.08838834764831845f * 1.44269504088896340736f)
#define NEG_INIT       -1e30f
#define MASK_VAL       -1e10f

union V16U { v16h v; h8 u[2]; _Float16 h[16]; };

__device__ __forceinline__ v8f wmma_f16(v16h a, v16h b, v8f c) {
    // D = A(16x32 f16) * B(32x16 f16) + C(16x16 f32)
    return __builtin_amdgcn_wmma_f32_16x16x32_f16(
        false, a, false, b, (short)0, c, false, false);
}

__global__ __launch_bounds__(THREADS) void
paged_prefill_attn_wmma(const float* __restrict__ q,
                        const float* __restrict__ k_cache,
                        const float* __restrict__ v_cache,
                        const int*   __restrict__ cu_seqlens_q,
                        const int*   __restrict__ cu_seqlens_k,
                        const int*   __restrict__ block_tables,
                        float*       __restrict__ out)
{
    __shared__ __align__(16) _Float16 Ks[KTILE * HEAD_DIM];      // [key][dim] 16KB
    __shared__ __align__(16) _Float16 Vt[HEAD_DIM * KTILE];      // [dim][key] 16KB
    __shared__ __align__(16) _Float16 Ps[NWAVES][16 * KTILE];    // P scratch  16KB

    const int tid   = threadIdx.x;
    const int wave  = tid >> 5;
    const int lane  = tid & 31;
    const int lhalf = lane & 15;   // N index within 16-wide group
    const int hi    = lane >> 4;   // 0: lanes 0-15, 1: lanes 16-31

    int bid = blockIdx.x;
    const int head  = bid % NUM_HEADS;          bid /= NUM_HEADS;
    const int qtile = bid % (LQ / QTILE);
    const int seq   = bid / (LQ / QTILE);
    const int kvh   = head / GQA_GROUP;

    const int q_start = cu_seqlens_q[seq];
    const int lq      = cu_seqlens_q[seq + 1] - q_start;
    const int lk      = cu_seqlens_k[seq + 1] - cu_seqlens_k[seq];
    const int hist    = lk - lq;

    const int q0   = qtile * QTILE;
    const int row0 = wave * 16;

    // ---- Q tile (16 x 128) as 4 A-fragments, scale*log2e folded in ----
    v16h Qf[4];
    {
        const int qrow = q_start + q0 + row0 + lhalf;       // A-layout: M = lane%16
        const float* qp = q + ((size_t)qrow * NUM_HEADS + head) * HEAD_DIM;
#pragma unroll
        for (int kc = 0; kc < 4; ++kc) {
            V16U t;
            const float* p0 = qp + kc * 32 + hi * 16;       // halves 0..7
            const float* p1 = qp + kc * 32 + hi * 16 + 8;   // halves 8..15
#pragma unroll
            for (int j = 0; j < 8; ++j) {
                t.h[j]     = (_Float16)(p0[j] * QSCALE_LOG2);
                t.h[8 + j] = (_Float16)(p1[j] * QSCALE_LOG2);
            }
            Qf[kc] = t.v;
        }
    }

    // ---- flash-attention state ----
    v8f O[8];
#pragma unroll
    for (int d = 0; d < 8; ++d)
#pragma unroll
        for (int r = 0; r < 8; ++r) O[d][r] = 0.0f;
    float m[8], ssum[8];
#pragma unroll
    for (int r = 0; r < 8; ++r) { m[r] = NEG_INIT; ssum[r] = 0.0f; }

    const int qpos_base = hist + q0;
    const int num_kt = (hist + q0 + QTILE + KTILE - 1) / KTILE;

    for (int kt = 0; kt < num_kt; ++kt) {
        __syncthreads();   // previous iteration's LDS reads complete

        // ---- cooperative stage: one physical block (64 keys x 128 dims) ----
        const int pb = block_tables[seq * MAX_NUM_BLOCKS + kt];  // KTILE==BLOCK_SIZE
#pragma unroll
        for (int it = tid; it < KTILE * (HEAD_DIM / 4); it += THREADS) {
            const int krow = it >> 5;            // HEAD_DIM/4 == 32
            const int dim  = (it & 31) * 4;
            const size_t base =
                (((size_t)pb * BLOCK_SIZE + krow) * NUM_KV_HEADS + kvh) * HEAD_DIM + dim;
            const float4 k4 = *(const float4*)(k_cache + base);
            h4 kh = { (_Float16)k4.x, (_Float16)k4.y, (_Float16)k4.z, (_Float16)k4.w };
            *(h4*)&Ks[krow * HEAD_DIM + dim] = kh;           // key-major
            const float4 v4 = *(const float4*)(v_cache + base);
            Vt[(dim + 0) * KTILE + krow] = (_Float16)v4.x;   // dim-major
            Vt[(dim + 1) * KTILE + krow] = (_Float16)v4.y;
            Vt[(dim + 2) * KTILE + krow] = (_Float16)v4.z;
            Vt[(dim + 3) * KTILE + krow] = (_Float16)v4.w;
        }
        // prefetch next physical block into L2 while this tile is consumed
        if (kt + 1 < num_kt) {
            const int pbn = block_tables[seq * MAX_NUM_BLOCKS + kt + 1];
            const int prow = tid & 63;
            const size_t pbase =
                (((size_t)pbn * BLOCK_SIZE + prow) * NUM_KV_HEADS + kvh) * HEAD_DIM
                + (size_t)(tid >> 6) * 32;      // 4 x 128B per 512B row
            __builtin_prefetch(k_cache + pbase, 0, 1);
            __builtin_prefetch(v_cache + pbase, 0, 1);
        }
        __syncthreads();

        // ---- S = Q * K^T : four 16x16 score tiles, double-buffered B frags ----
        v8f c[4];
#pragma unroll
        for (int n = 0; n < 4; ++n)
#pragma unroll
            for (int r = 0; r < 8; ++r) c[n][r] = 0.0f;

#pragma unroll
        for (int n = 0; n < 4; ++n) {
            const _Float16* kp = &Ks[(n * 16 + lhalf) * HEAD_DIM + hi * 16];
            V16U b0, b1;
            b0.u[0] = *(const h8*)kp;  b0.u[1] = *(const h8*)(kp + 8);
#pragma unroll
            for (int kc = 0; kc < 4; ++kc) {
                V16U* nb = (kc & 1) ? &b0 : &b1;     // load next while computing
                if (kc < 3) {
                    const _Float16* p = kp + (kc + 1) * 32;
                    nb->u[0] = *(const h8*)p;  nb->u[1] = *(const h8*)(p + 8);
                }
                c[n] = wmma_f16(Qf[kc], (kc & 1) ? b1.v : b0.v, c[n]);
            }
        }

        // ---- causal mask: wave-uniform fast path skips all mask VALU ----
        const int kbase = kt * KTILE;
        if (kbase + (KTILE - 1) > qpos_base + row0) {        // scalar branch
#pragma unroll
            for (int n = 0; n < 4; ++n) {
                const int kpos = kbase + n * 16 + lhalf;
#pragma unroll
                for (int r = 0; r < 8; ++r) {
                    const int qpos = qpos_base + row0 + r + hi * 8;
                    if (kpos > qpos) c[n][r] = MASK_VAL;
                }
            }
        }

        // ---- online softmax (exp2 domain) ----
#pragma unroll
        for (int r = 0; r < 8; ++r) {
            float mx = fmaxf(fmaxf(c[0][r], c[1][r]), fmaxf(c[2][r], c[3][r]));
#pragma unroll
            for (int off = 8; off > 0; off >>= 1)
                mx = fmaxf(mx, __shfl_xor(mx, off, 32));
            mx = fmaxf(mx, m[r]);
            const float alpha = exp2f(m[r] - mx);
            m[r] = mx;

            float rs = 0.0f;
#pragma unroll
            for (int n = 0; n < 4; ++n) {
                const float p = exp2f(c[n][r] - mx);   // masked -> 0
                c[n][r] = p;
                rs += p;
            }
#pragma unroll
            for (int off = 8; off > 0; off >>= 1)
                rs += __shfl_xor(rs, off, 32);
            ssum[r] = ssum[r] * alpha + rs;
#pragma unroll
            for (int d = 0; d < 8; ++d) O[d][r] *= alpha;
        }

        // ---- C-layout -> A-layout for P via per-wave LDS scratch ----
#pragma unroll
        for (int n = 0; n < 4; ++n)
#pragma unroll
            for (int r = 0; r < 8; ++r) {
                const int M = r + hi * 8;
                Ps[wave][M * KTILE + n * 16 + lhalf] = (_Float16)c[n][r];
            }
        __syncthreads();

        v16h Pf[2];
#pragma unroll
        for (int j = 0; j < 2; ++j) {
            const _Float16* pp = &Ps[wave][lhalf * KTILE + j * 32 + hi * 16];
            V16U pa;
            pa.u[0] = *(const h8*)pp;        // keys j*32 + hi*16 .. +7
            pa.u[1] = *(const h8*)(pp + 8);  // keys j*32 + hi*16 + 8 .. +15
            Pf[j] = pa.v;
        }

        // ---- O += P * V (B frags contiguous thanks to transposed Vt) ----
#pragma unroll
        for (int d = 0; d < 8; ++d) {
            const _Float16* vp = &Vt[(d * 16 + lhalf) * KTILE + hi * 16];
            V16U v0, v1;
            v0.u[0] = *(const h8*)vp;         v0.u[1] = *(const h8*)(vp + 8);
            v1.u[0] = *(const h8*)(vp + 32);  v1.u[1] = *(const h8*)(vp + 40);
            O[d] = wmma_f16(Pf[0], v0.v, O[d]);
            O[d] = wmma_f16(Pf[1], v1.v, O[d]);
        }
    }

    // ---- normalize and write out ----
    const int token0 = q_start + q0 + row0;
#pragma unroll
    for (int r = 0; r < 8; ++r) {
        const float inv = 1.0f / ssum[r];
        const size_t obase =
            ((size_t)(token0 + r + hi * 8) * NUM_HEADS + head) * HEAD_DIM + lhalf;
#pragma unroll
        for (int d = 0; d < 8; ++d)
            out[obase + d * 16] = O[d][r] * inv;
    }
}

extern "C" void kernel_launch(void* const* d_in, const int* in_sizes, int n_in,
                              void* d_out, int out_size, void* d_ws, size_t ws_size,
                              hipStream_t stream) {
    (void)in_sizes; (void)n_in; (void)out_size; (void)d_ws; (void)ws_size;
    const float* q       = (const float*)d_in[0];
    const float* k_cache = (const float*)d_in[1];
    const float* v_cache = (const float*)d_in[2];
    const int*   cu_q    = (const int*)d_in[3];
    const int*   cu_k    = (const int*)d_in[4];
    const int*   btables = (const int*)d_in[5];
    float*       out     = (float*)d_out;

    dim3 grid(NUM_SEQS * (LQ / QTILE) * NUM_HEADS);   // 2 * 8 * 32 = 512 WGs
    dim3 block(THREADS);                              // 8 waves (wave32)
    paged_prefill_attn_wmma<<<grid, block, 0, stream>>>(
        q, k_cache, v_cache, cu_q, cu_k, btables, out);
}
